// GINModel_5153960755352
// MI455X (gfx1250) — matrix operand order repaired and checked by reading, not compiled, not run on hardware
//
#include <hip/hip_runtime.h>

#define NNODES 50000
#define NEDGES 600000
#define DIMF   128
#define OUTD   40
#define OUTP   48   // padded head width (3 tiles of 16)

typedef __attribute__((ext_vector_type(16))) _Float16 v16h;
typedef __attribute__((ext_vector_type(8)))  _Float16 v8h;
typedef __attribute__((ext_vector_type(4)))  _Float16 v4h;
typedef __attribute__((ext_vector_type(8)))  float    v8f;

__device__ __forceinline__ void atomAddF32(float* p, float v) {
    (void)__hip_atomic_fetch_add(p, v, __ATOMIC_RELAXED, __HIP_MEMORY_SCOPE_AGENT);
}

// ---------------- elementwise helpers ----------------

__global__ void k_copy_f32(const float* __restrict__ in, float* __restrict__ out, int n4) {
    int t = blockIdx.x * blockDim.x + threadIdx.x;
    if (t < n4) ((float4*)out)[t] = ((const float4*)in)[t];
}

__global__ void k_cvt_f32_to_f16(const float* __restrict__ in, _Float16* __restrict__ out, int n4) {
    int t = blockIdx.x * blockDim.x + threadIdx.x;
    if (t >= n4) return;
    float4 v = ((const float4*)in)[t];
    v4h h; h[0] = (_Float16)v.x; h[1] = (_Float16)v.y; h[2] = (_Float16)v.z; h[3] = (_Float16)v.w;
    ((v4h*)out)[t] = h;
}

__global__ void k_cvt_f16_to_f32(const _Float16* __restrict__ in, float* __restrict__ out, int n4) {
    int t = blockIdx.x * blockDim.x + threadIdx.x;
    if (t >= n4) return;
    v4h h = ((const v4h*)in)[t];
    float4 v; v.x = (float)h[0]; v.y = (float)h[1]; v.z = (float)h[2]; v.w = (float)h[3];
    ((float4*)out)[t] = v;
}

// Pre-swizzle weights into B-fragment-major order (f32 -> f16):
// out[((tile*4 + chunk)*32 + lane)*16 + j] = W[(chunk*32 + (j>=8)*16 + (lane>=16)*8 + (j&7)) * ncols_in + tile*16 + (lane&15)]
// (zero-padded when the column exceeds ncols_in, e.g. head 40 -> 48)
__global__ void k_prep_w(const float* __restrict__ in, _Float16* __restrict__ out,
                         int ncols_in, int colTiles) {
    int t = blockIdx.x * blockDim.x + threadIdx.x;
    if (t >= colTiles * 2048) return;            // 4 chunks * 32 lanes * 16 halves per tile
    int j     = t & 15;
    int lane  = (t >> 4) & 31;
    int chunk = (t >> 9) & 3;
    int tile  = t >> 11;
    int k = chunk * 32 + ((j >> 3) << 4) + ((lane >> 4) << 3) + (j & 7);
    int n = (tile << 4) + (lane & 15);
    out[t] = (n < ncols_in) ? (_Float16)in[k * ncols_in + n] : (_Float16)0.f;
}

// ---------------- edge scatter (GIN sum aggregation) ----------------
// one wave32 per edge: 32 lanes x float4 = full 128-float row

__global__ void k_scatter_f32(const float* __restrict__ feat, const int* __restrict__ ei,
                              float* __restrict__ agg) {
    long t = (long)blockIdx.x * blockDim.x + threadIdx.x;
    if (t >= (long)NEDGES * 32) return;
    int e = (int)(t >> 5);
    int c = (int)(t & 31) << 2;
    int s = ei[e], d = ei[NEDGES + e];
    float4 v = *(const float4*)(feat + s * DIMF + c);
    float* p = agg + d * DIMF + c;
    atomAddF32(p + 0, v.x); atomAddF32(p + 1, v.y);
    atomAddF32(p + 2, v.z); atomAddF32(p + 3, v.w);
}

__global__ void k_scatter_f16(const _Float16* __restrict__ feat, const int* __restrict__ ei,
                              float* __restrict__ agg) {
    long t = (long)blockIdx.x * blockDim.x + threadIdx.x;
    if (t >= (long)NEDGES * 32) return;
    int e = (int)(t >> 5);
    int c = (int)(t & 31) << 2;
    int s = ei[e], d = ei[NEDGES + e];
    v4h h = *(const v4h*)(feat + s * DIMF + c);
    float* p = agg + d * DIMF + c;
    atomAddF32(p + 0, (float)h[0]); atomAddF32(p + 1, (float)h[1]);
    atomAddF32(p + 2, (float)h[2]); atomAddF32(p + 3, (float)h[3]);
}

// ---------------- WMMA GEMM: out = act(A[50000 x 128] @ W[128 x *] + bias) ----------------
// block = 128 threads (4 waves); wave -> one 16x16 output tile (colTile = blockIdx.y*4 + wave)
// A: f16 row-major stride DIMF -> two aligned v8h loads per K-chunk (ISA 16-bit A layout).
// Wsw: pre-swizzled fragment-major f16 -> two aligned v8h loads per K-chunk. Accum f32.

template <bool RELU, bool OUTH>
__global__ void k_gemm_wmma(const _Float16* __restrict__ A, const _Float16* __restrict__ Wsw,
                            const float* __restrict__ bias, void* __restrict__ outp,
                            int ldo, int ncols, int colTiles) {
    int wave = threadIdx.x >> 5;
    int lane = threadIdx.x & 31;
    int colTile = blockIdx.y * 4 + wave;
    if (colTile >= colTiles) return;                  // wave-uniform: EXEC stays all-ones
    int n0 = colTile << 4;
    int m0 = blockIdx.x << 4;

    int col = n0 + (lane & 15);
    float bval = (col < ncols) ? bias[col] : 0.f;
    v8f c;
#pragma unroll
    for (int r = 0; r < 8; ++r) c[r] = bval;

    const _Float16* Arow = A + (m0 + (lane & 15)) * DIMF + ((lane >> 4) << 3);
    const v8h* Wf = (const v8h*)Wsw + (((colTile << 2) << 5) + lane) * 2;

#pragma unroll
    for (int kc = 0; kc < 4; ++kc) {
        union { v16h v; v8h h[2]; } a, b;
        a.h[0] = *(const v8h*)(Arow + kc * 32);
        a.h[1] = *(const v8h*)(Arow + kc * 32 + 16);
        b.h[0] = Wf[kc * 64 + 0];
        b.h[1] = Wf[kc * 64 + 1];
        c = __builtin_amdgcn_wmma_f32_16x16x32_f16(false, a.v, false, b.v, (short)0, c,
                                                   false, false);
    }

    int rbase = m0 + ((lane >> 4) << 3);              // rows: VGPR r -> rbase + r
    if (OUTH) {
        _Float16* out = (_Float16*)outp;
#pragma unroll
        for (int r = 0; r < 8; ++r) {
            float v = c[r];
            if (RELU) v = v > 0.f ? v : 0.f;
            out[(rbase + r) * ldo + col] = (_Float16)v;
        }
    } else {
        float* out = (float*)outp;
        if (col < ncols) {
#pragma unroll
            for (int r = 0; r < 8; ++r) {
                float v = c[r];
                if (RELU) v = v > 0.f ? v : 0.f;
                out[(rbase + r) * ldo + col] = v;
            }
        }
    }
}

// ---------------- launcher ----------------

extern "C" void kernel_launch(void* const* d_in, const int* in_sizes, int n_in,
                              void* d_out, int out_size, void* d_ws, size_t ws_size,
                              hipStream_t stream) {
    const float* x   = (const float*)d_in[0];
    const int*   ei  = (const int*)d_in[1];
    const float* W1a = (const float*)d_in[2];  const float* b1a = (const float*)d_in[3];
    const float* W1b = (const float*)d_in[4];  const float* b1b = (const float*)d_in[5];
    const float* W2a = (const float*)d_in[6];  const float* b2a = (const float*)d_in[7];
    const float* W2b = (const float*)d_in[8];  const float* b2b = (const float*)d_in[9];
    const float* Wfc = (const float*)d_in[10]; const float* bfc = (const float*)d_in[11];
    float* out = (float*)d_out;

    const size_t AGG_B = (size_t)NNODES * DIMF * 4;   // 25,600,000
    const size_t F16_B = (size_t)NNODES * DIMF * 2;   // 12,800,000
    const size_t W_B   = (size_t)DIMF * DIMF * 2;     // 32,768 (8 tiles * 2048 halves * 2B)
    if (ws_size < AGG_B + 2 * F16_B + 4 * W_B + (size_t)DIMF * OUTP * 2) return;

    char* ws = (char*)d_ws;
    float*    agg   = (float*)ws;
    _Float16* Ah    = (_Float16*)(ws + AGG_B);
    _Float16* Bh    = (_Float16*)(ws + AGG_B + F16_B);
    _Float16* W1a_h = (_Float16*)(ws + AGG_B + 2 * F16_B);
    _Float16* W1b_h = W1a_h + DIMF * DIMF;
    _Float16* W2a_h = W1b_h + DIMF * DIMF;
    _Float16* W2b_h = W2a_h + DIMF * DIMF;
    _Float16* Wfc_h = W2b_h + DIMF * DIMF;            // 3 tiles * 2048 halves

    // weight prep: f32 -> f16 in B-fragment-major (swizzled) order
    k_prep_w<<<64, 256, 0, stream>>>(W1a, W1a_h, DIMF, 8);
    k_prep_w<<<64, 256, 0, stream>>>(W1b, W1b_h, DIMF, 8);
    k_prep_w<<<64, 256, 0, stream>>>(W2a, W2a_h, DIMF, 8);
    k_prep_w<<<64, 256, 0, stream>>>(W2b, W2b_h, DIMF, 8);
    k_prep_w<<<24, 256, 0, stream>>>(Wfc, Wfc_h, OUTD, 3);

    const int n4 = NNODES * DIMF / 4;                 // 1.6M float4s
    const int nb = (n4 + 255) / 256;
    const int eb = (int)(((long)NEDGES * 32 + 255) / 256);
    dim3 blk(128);
    dim3 gSq(NNODES / 16, 2);   // 8 col tiles / 4 waves
    dim3 gHd(NNODES / 16, 1);   // 3 col tiles

    // ---- layer 1 ----
    k_copy_f32<<<nb, 256, 0, stream>>>(x, agg, n4);                 // agg = x
    k_scatter_f32<<<eb, 256, 0, stream>>>(x, ei, agg);              // agg += sum x[src]
    k_cvt_f32_to_f16<<<nb, 256, 0, stream>>>(agg, Ah, n4);
    k_gemm_wmma<true,  true><<<gSq, blk, 0, stream>>>(Ah, W1a_h, b1a, Bh, DIMF, DIMF, 8);
    k_gemm_wmma<true,  true><<<gSq, blk, 0, stream>>>(Bh, W1b_h, b1b, Ah, DIMF, DIMF, 8);
    // Ah now holds h1 = relu(gin_conv1(x))  (outer relu fused)

    // ---- layer 2 ----
    k_cvt_f16_to_f32<<<nb, 256, 0, stream>>>(Ah, agg, n4);          // agg = h1
    k_scatter_f16<<<eb, 256, 0, stream>>>(Ah, ei, agg);             // agg += sum h1[src]
    k_cvt_f32_to_f16<<<nb, 256, 0, stream>>>(agg, Bh, n4);
    k_gemm_wmma<true,  true><<<gSq, blk, 0, stream>>>(Bh, W2a_h, b2a, Ah, DIMF, DIMF, 8);
    k_gemm_wmma<false, true><<<gSq, blk, 0, stream>>>(Ah, W2b_h, b2b, Bh, DIMF, DIMF, 8);
    // Bh now holds h2 = gin_conv2(h1) (no relu)

    // ---- head: out[N x 40] = h2 @ Wfc + bfc ----
    k_gemm_wmma<false, false><<<gHd, blk, 0, stream>>>(Bh, Wfc_h, bfc, out, OUTD, OUTD, 3);
}
